// WeightedConvTranspose_23785528886125
// MI455X (gfx1250) — matrix operand order repaired and checked by reading, not compiled, not run on hardware
//
#include <hip/hip_runtime.h>
#include <math.h>

// ---------------------------------------------------------------------------
// WeightedConvTranspose on MI455X (gfx1250, wave32, WMMA).
// out[b,o,l] = sum_{c,k} g[b,k,l] * x_t[b,c,l+k-4] * W[o,c,k]
// Zero-insert upsample => even l uses even taps (K=320), odd l odd taps (K=256):
// two bf16 WMMA GEMMs per tile, f32 accumulation.
// ---------------------------------------------------------------------------

typedef __bf16 bf16;
typedef bf16  v2bf  __attribute__((ext_vector_type(2)));
typedef bf16  v8bf  __attribute__((ext_vector_type(8)));
typedef bf16  v16bf __attribute__((ext_vector_type(16)));
typedef float v8f   __attribute__((ext_vector_type(8)));

#define BATCH 8
#define CIN   64
#define COUT  64
#define NX    16384      // input curve length
#define LL    32768      // output curve length (stride*NX)
#define TL    128        // output columns per workgroup
#define SA    584        // A row stride in bf16 (576 + 8 pad)
#define SBE   328        // even-phase B row stride (320 + 8 pad)
#define SBO   264        // odd-phase  B row stride (256 + 8 pad)

// LDS layout (byte offsets, all 16B aligned)
#define A_OFF 0                      // bf16 A[64][SA]          = 74752 B
#define B_OFF 74752                  // bf16 B[64][SBE] (shared even/odd) = 41984 B
#define X_OFF 116736                 // bf16 X[68][66]          = 8976 B
#define C_OFF 125712                 // f32  C[3][136]          = 1632 B
#define G_OFF 127344                 // f32  G[9][128]          = 4608 B
#define SMEM_BYTES 131952

static __device__ __forceinline__ v16bf cat16(v8bf a, v8bf b) {
  return __builtin_shufflevector(a, b, 0,1,2,3,4,5,6,7,8,9,10,11,12,13,14,15);
}

__global__ __launch_bounds__(256)
void wct_kernel(const float* __restrict__ x, const float* __restrict__ coords,
                const float* __restrict__ sigma, const float* __restrict__ w,
                float* __restrict__ out)
{
  extern __shared__ unsigned char smem[];
  bf16*  sA = (bf16*)(smem + A_OFF);
  bf16*  sB = (bf16*)(smem + B_OFF);
  bf16*  sX = (bf16*)(smem + X_OFF);
  float* sC = (float*)(smem + C_OFF);
  float* sG = (float*)(smem + G_OFF);

  const int tid = threadIdx.x;
  const int b   = blockIdx.x >> 8;            // 256 tiles per batch
  const int l0  = (blockIdx.x & 255) * TL;

  // ---- stage coords window: sC[d][i] = coords[b,d,l0+i-4] (zero-padded) ----
  for (int t = tid; t < 3 * 136; t += 256) {
    int d = t / 136, i = t - d * 136;
    int gi = l0 + i - 4;
    sC[t] = (gi >= 0 && gi < LL) ? coords[((size_t)b * 3 + d) * LL + gi] : 0.0f;
  }
  __syncthreads();

  const float rsig = 1.0f / sigma[0];

  // ---- geometric weights g[j][lt] ----
  for (int t = tid; t < 9 * 128; t += 256) {
    int j = t >> 7, lt = t & 127;
    float dx = sC[lt + j]       - sC[lt + 4];
    float dy = sC[136 + lt + j] - sC[136 + lt + 4];
    float dz = sC[272 + lt + j] - sC[272 + lt + 4];
    float s  = sqrtf(dx * dx + dy * dy + dz * dz);
    sG[t] = fmaxf(1.0f - s * rsig, 0.0f);
  }

  // ---- x tile: sX[i][c] = x[b,c,l0/2 - 2 + i], i in [0,68) ----
  {
    const int xbase = (l0 >> 1) - 2;
    for (int t = tid; t < 64 * 68; t += 256) {
      int c = t / 68, i = t - c * 68;
      int gi = xbase + i;
      float v = (gi >= 0 && gi < NX) ? x[((size_t)b * 64 + c) * NX + gi] : 0.0f;
      sX[i * 66 + c] = (bf16)v;
    }
  }

  // ---- weight matrix A: sA[o][kt*64 + c] = W[o,c,kt] (bf16) ----
  for (int t = tid; t < 64 * 576; t += 256) {
    int o = t / 576, r = t - o * 576;
    int c = r / 9,  kt = r - c * 9;
    sA[o * SA + kt * 64 + c] = (bf16)w[t];
  }
  __syncthreads();

  // ---- build B_even: B[n][j'*64+c] = g[2j'][2n] * x[c][n+j'] ----
  for (int e = tid; e < 64 * 5 * 32; e += 256) {     // bf16 pairs
    int n  = e / 160;
    int r  = e - n * 160;
    int j2 = r >> 5;
    int cp = (r & 31) << 1;
    float gv = sG[(2 * j2) * 128 + 2 * n];
    v2bf xv = *(const v2bf*)(sX + (n + j2) * 66 + cp);
    v2bf pv;
    pv.x = (bf16)((float)xv.x * gv);
    pv.y = (bf16)((float)xv.y * gv);
    *(v2bf*)(sB + n * SBE + j2 * 64 + cp) = pv;
  }
  __syncthreads();

  // ---- wave tiling: 8 waves, each one 16x16 M-tile x two N-tiles ----
  const int wv = tid >> 5, lane = tid & 31;
  const int lm = lane & 15, lh = lane >> 4;
  const int m0 = (wv & 3) * 16;
  const int n0 = (wv >> 2) * 32;

  // ======================= EVEN columns: K = 320 ==========================
  {
    v8f acc0 = {}, acc1 = {};
    const bf16* pa_base = sA + (m0 + lm) * SA + lh * 16;
    const bf16* pb_base = sB + (n0 + lm) * SBE + lh * 16;
#pragma unroll
    for (int s = 0; s < 10; ++s) {
      const bf16* pa = pa_base + ((s >> 1) * 128 + (s & 1) * 32);
      v16bf av = cat16(*(const v8bf*)pa, *(const v8bf*)(pa + 8));
      const bf16* pb = pb_base + s * 32;
      v16bf bv0 = cat16(*(const v8bf*)pb,
                        *(const v8bf*)(pb + 8));
      v16bf bv1 = cat16(*(const v8bf*)(pb + 16 * SBE),
                        *(const v8bf*)(pb + 16 * SBE + 8));
      acc0 = __builtin_amdgcn_wmma_f32_16x16x32_bf16(false, av, false, bv0,
                                                     (short)0, acc0, false, false);
      acc1 = __builtin_amdgcn_wmma_f32_16x16x32_bf16(false, av, false, bv1,
                                                     (short)0, acc1, false, false);
    }
    const size_t obase = ((size_t)b * 64 + m0 + 8 * lh) * (size_t)LL;
    const int col0 = l0 + 2 * (n0 + lm);
#pragma unroll
    for (int r = 0; r < 8; ++r) {
      out[obase + (size_t)r * LL + col0]      = acc0[r];
      out[obase + (size_t)r * LL + col0 + 32] = acc1[r];
    }
  }
  __syncthreads();   // everyone done reading sB before odd rebuild

  // ---- build B_odd: B[n][j''*64+c] = g[2j''+1][2n+1] * x[c][n+j''+1] ----
  for (int e = tid; e < 64 * 4 * 32; e += 256) {
    int n  = e >> 7;
    int r  = e & 127;
    int j2 = r >> 5;
    int cp = (r & 31) << 1;
    float gv = sG[(2 * j2 + 1) * 128 + 2 * n + 1];
    v2bf xv = *(const v2bf*)(sX + (n + j2 + 1) * 66 + cp);
    v2bf pv;
    pv.x = (bf16)((float)xv.x * gv);
    pv.y = (bf16)((float)xv.y * gv);
    *(v2bf*)(sB + n * SBO + j2 * 64 + cp) = pv;
  }
  __syncthreads();

  // ======================= ODD columns: K = 256 ===========================
  {
    v8f acc0 = {}, acc1 = {};
    const bf16* pa_base = sA + (m0 + lm) * SA + lh * 16;
    const bf16* pb_base = sB + (n0 + lm) * SBO + lh * 16;
#pragma unroll
    for (int s = 0; s < 8; ++s) {
      const bf16* pa = pa_base + ((s >> 1) * 128 + 64 + (s & 1) * 32);
      v16bf av = cat16(*(const v8bf*)pa, *(const v8bf*)(pa + 8));
      const bf16* pb = pb_base + s * 32;
      v16bf bv0 = cat16(*(const v8bf*)pb,
                        *(const v8bf*)(pb + 8));
      v16bf bv1 = cat16(*(const v8bf*)(pb + 16 * SBO),
                        *(const v8bf*)(pb + 16 * SBO + 8));
      acc0 = __builtin_amdgcn_wmma_f32_16x16x32_bf16(false, av, false, bv0,
                                                     (short)0, acc0, false, false);
      acc1 = __builtin_amdgcn_wmma_f32_16x16x32_bf16(false, av, false, bv1,
                                                     (short)0, acc1, false, false);
    }
    const size_t obase = ((size_t)b * 64 + m0 + 8 * lh) * (size_t)LL;
    const int col0 = l0 + 2 * (n0 + lm) + 1;
#pragma unroll
    for (int r = 0; r < 8; ++r) {
      out[obase + (size_t)r * LL + col0]      = acc0[r];
      out[obase + (size_t)r * LL + col0 + 32] = acc1[r];
    }
  }
}

extern "C" void kernel_launch(void* const* d_in, const int* in_sizes, int n_in,
                              void* d_out, int out_size, void* d_ws, size_t ws_size,
                              hipStream_t stream) {
  (void)in_sizes; (void)n_in; (void)out_size; (void)d_ws; (void)ws_size;
  const float* x      = (const float*)d_in[0];
  const float* coords = (const float*)d_in[1];
  const float* sigma  = (const float*)d_in[2];
  const float* w      = (const float*)d_in[3];
  float* out = (float*)d_out;

  dim3 grid(BATCH * (LL / TL));   // 8 * 256 = 2048 workgroups
  dim3 block(256);                // 8 wave32 waves
  wct_kernel<<<grid, block, SMEM_BYTES, stream>>>(x, coords, sigma, w, out);
}
